// _QuantModel_46222438039671
// MI455X (gfx1250) — matrix operand order repaired and checked by reading, not compiled, not run on hardware
//
#include <hip/hip_runtime.h>

// ---------------------------------------------------------------------------
// MXFP4 two-layer quantized linear for MI455X (gfx1250, wave32, WMMA).
//   h = x @ dequant(W1)^T + b1 ;  y = h @ dequant(W2)^T + b2
// v3: 256x128 block tile (512 thr = 16 waves, wave tile 64x32), BK=64,
//     double-buffered LDS (96KB) filled with global_load_async_to_lds_b128
//     (ASYNCcnt) overlapping v_wmma_f32_16x16x32_bf16 compute.
// Weights dequantize EXACTLY to bf16 (E2M1 * 2^e); x->bf16 once; both W
// (32MB) and x (64MB) stay resident in the 192MB L2 across the block grid.
// ---------------------------------------------------------------------------

typedef __attribute__((ext_vector_type(16))) __bf16 v16bf;
typedef __attribute__((ext_vector_type(8)))  float  v8f;

static constexpr int TDIM = 8192;
static constexpr int DDIM = 4096;

__device__ __constant__ float FP4LUT[16] = {
    0.0f, 0.5f, 1.0f, 1.5f, 2.0f, 3.0f, 4.0f, 6.0f,
   -0.0f,-0.5f,-1.0f,-1.5f,-2.0f,-3.0f,-4.0f,-6.0f};

// ---- async global->LDS 16B copy, per lane (ASYNCcnt tracked) --------------
__device__ __forceinline__ void async_copy16(const void* gptr, void* lds_ptr) {
    unsigned lds = (unsigned)(size_t)lds_ptr;              // low 32b = LDS offset
    unsigned long long g = (unsigned long long)(size_t)gptr;
    asm volatile("global_load_async_to_lds_b128 %0, %1, off"
                 :: "v"(lds), "v"(g) : "memory");
}
__device__ __forceinline__ void wait_async_le6() {
    asm volatile("s_wait_asynccnt 0x6" ::: "memory");
}
__device__ __forceinline__ void wait_async_le0() {
    asm volatile("s_wait_asynccnt 0x0" ::: "memory");
}

// ---- W dequant: codes[D,D] (int, 0..15) * scales[D,D/32] -> bf16 [D,D] ----
__global__ __launch_bounds__(256) void dequant_w_kernel(
    const int* __restrict__ codes, const float* __restrict__ scales,
    __bf16* __restrict__ out)
{
    size_t idx = (size_t)blockIdx.x * 256 + threadIdx.x;
    int col = (int)(idx & (DDIM - 1));
    int row = (int)(idx >> 12);                            // DDIM == 4096
    float s = scales[(size_t)row * (DDIM / 32) + (col >> 5)];
    int   c = codes[idx] & 15;
    out[idx] = (__bf16)(FP4LUT[c] * s);                    // exact in bf16
}

// ---- x f32 -> bf16 ----
__global__ __launch_bounds__(256) void f32_to_bf16_kernel(
    const float* __restrict__ in, __bf16* __restrict__ out)
{
    size_t i = (size_t)blockIdx.x * 256 + threadIdx.x;
    out[i] = (__bf16)in[i];
}

// ---- tiled GEMM: C[M,N] = A[M,K] @ B[N,K]^T + bias, bf16 in, f32 accum ----
// block tile 256x128, 512 threads = 16 waves (4x4), wave tile 64x32.
// K-tile 64, double-buffered LDS filled by async DMA overlapping WMMA.
__global__ __launch_bounds__(512) void gemm_bf16_kernel(
    const __bf16* __restrict__ A,     // [M,K] row-major
    const __bf16* __restrict__ B,     // [N,K] row-major
    const float*  __restrict__ bias,  // [N]
    __bf16* __restrict__ out_bf16,    // layer-1 target (or null)
    float*  __restrict__ out_f32,     // layer-2 target (or null)
    int M, int N, int K)
{
    __shared__ __bf16 As[2][256 * 64];   // 2 x 32 KB
    __shared__ __bf16 Bs[2][128 * 64];   // 2 x 16 KB

    const int tid  = threadIdx.x;
    const int lane = tid & 31;
    const int wave = tid >> 5;        // 0..15
    const int wm   = wave & 3;        // 0..3 -> 64-row slab of 256
    const int wn   = wave >> 2;       // 0..3 -> 32-col slab of 128
    const int h    = lane >> 4;       // lane half (ISA frag layout)
    const int l16  = lane & 15;

    const int rowBase = blockIdx.y * 256;
    const int colBase = blockIdx.x * 128;

    v8f zero = {};
    v8f acc[4][2];
#pragma unroll
    for (int i = 0; i < 4; ++i)
#pragma unroll
        for (int j = 0; j < 2; ++j) acc[i][j] = zero;

    // async fill of one 256x64 A tile + 128x64 B tile (4+2 ops/thread)
    auto fill = [&](int buf, int k0) {
#pragma unroll
        for (int j = 0; j < 4; ++j) {
            int chunk = j * 512 + tid;        // 0..2047
            int r  = chunk >> 3;              // 0..255 tile row
            int ck = (chunk & 7) * 8;         // bf16 col offset (16B chunks)
            async_copy16(A + (size_t)(rowBase + r) * K + k0 + ck,
                         &As[buf][r * 64 + ck]);
        }
#pragma unroll
        for (int j = 0; j < 2; ++j) {
            int chunk = j * 512 + tid;        // 0..1023
            int r  = chunk >> 3;              // 0..127 tile row
            int ck = (chunk & 7) * 8;
            async_copy16(B + (size_t)(colBase + r) * K + k0 + ck,
                         &Bs[buf][r * 64 + ck]);
        }
    };

    const int nk = K / 64;
    fill(0, 0);                               // prologue: tile 0 in flight

    for (int kt = 0; kt < nk; ++kt) {
        const int cur = kt & 1;
        if (kt + 1 < nk) {                    // overlap: issue next tile
            fill(cur ^ 1, (kt + 1) * 64);
            wait_async_le6();                 // in-order: current tile landed
        } else {
            wait_async_le0();
        }
        __syncthreads();                      // all lanes' DMA visible

#pragma unroll
        for (int kk = 0; kk < 2; ++kk) {      // two 16x16x32 K-slabs
            v16bf a[4], b[2];
#pragma unroll
            for (int mi = 0; mi < 4; ++mi) {
                const __bf16* p =
                    &As[cur][(wm * 64 + mi * 16 + l16) * 64 + kk * 32 + h * 8];
                ((uint4*)&a[mi])[0] = *(const uint4*)(p);
                ((uint4*)&a[mi])[1] = *(const uint4*)(p + 16);
            }
#pragma unroll
            for (int ni = 0; ni < 2; ++ni) {
                const __bf16* q =
                    &Bs[cur][(wn * 32 + ni * 16 + l16) * 64 + kk * 32 + h * 8];
                ((uint4*)&b[ni])[0] = *(const uint4*)(q);
                ((uint4*)&b[ni])[1] = *(const uint4*)(q + 16);
            }
#pragma unroll
            for (int mi = 0; mi < 4; ++mi)
#pragma unroll
                for (int ni = 0; ni < 2; ++ni)
                    acc[mi][ni] = __builtin_amdgcn_wmma_f32_16x16x32_bf16(
                        false, a[mi], false, b[ni],
                        (short)0, acc[mi][ni], false, false);
        }
        __syncthreads();                      // done reading buf before reuse
    }

    // epilogue: C layout -> row = r + h*8 (within 16), col = l16
#pragma unroll
    for (int mi = 0; mi < 4; ++mi) {
#pragma unroll
        for (int ni = 0; ni < 2; ++ni) {
            int col = colBase + wn * 32 + ni * 16 + l16;
            float bv = bias[col];
#pragma unroll
            for (int r = 0; r < 8; ++r) {
                int row = rowBase + wm * 64 + mi * 16 + h * 8 + r;
                float v = acc[mi][ni][r] + bv;
                if (out_f32) out_f32[(size_t)row * N + col] = v;
                else         out_bf16[(size_t)row * N + col] = (__bf16)v;
            }
        }
    }
}

extern "C" void kernel_launch(void* const* d_in, const int* in_sizes, int n_in,
                              void* d_out, int out_size, void* d_ws, size_t ws_size,
                              hipStream_t stream) {
    const float* x   = (const float*)d_in[0];
    const int*   w1c = (const int*)  d_in[1];
    const float* w1s = (const float*)d_in[2];
    const float* b1  = (const float*)d_in[3];
    const int*   w2c = (const int*)  d_in[4];
    const float* w2s = (const float*)d_in[5];
    const float* b2  = (const float*)d_in[6];
    float* y = (float*)d_out;

    // ws: x_bf16 (64MB) | h_bf16 (64MB) | w1_bf16 (32MB) | w2_bf16 (32MB)
    char* ws = (char*)d_ws;
    __bf16* x_bf  = (__bf16*)ws;
    __bf16* h_bf  = (__bf16*)(ws + (size_t)TDIM * DDIM * 2);
    __bf16* w1_bf = (__bf16*)(ws + (size_t)TDIM * DDIM * 4);
    __bf16* w2_bf = (__bf16*)(ws + (size_t)TDIM * DDIM * 4 + (size_t)DDIM * DDIM * 2);

    f32_to_bf16_kernel<<<(TDIM * (size_t)DDIM) / 256, 256, 0, stream>>>(x, x_bf);
    dequant_w_kernel<<<((size_t)DDIM * DDIM) / 256, 256, 0, stream>>>(w1c, w1s, w1_bf);
    dequant_w_kernel<<<((size_t)DDIM * DDIM) / 256, 256, 0, stream>>>(w2c, w2s, w2_bf);

    dim3 grid(DDIM / 128, TDIM / 256);   // (32, 32)
    gemm_bf16_kernel<<<grid, 512, 0, stream>>>(x_bf, w1_bf, b1, h_bf, nullptr,
                                               TDIM, DDIM, DDIM);
    gemm_bf16_kernel<<<grid, 512, 0, stream>>>(h_bf, w2_bf, b2, nullptr, y,
                                               TDIM, DDIM, DDIM);
}
